// PsiQRHTransformerBlock_81913616269398
// MI455X (gfx1250) — compile-verified, matrix operand order checked by reading, not run on hardware
//
#include <hip/hip_runtime.h>
#include <hip/hip_bf16.h>

#define BB 8
#define SS 4096
#define QDIM 4
#define DMM 1024
#define HH 16
#define HDD 64
#define NROT 4
#define NFR (SS * DMM)      // 4194304 = 2^22
#define NN1 2048
#define LGN1 11
#define LGS 12
#define ROWS (BB * SS)      // 32768

typedef __attribute__((ext_vector_type(16))) __bf16 v16bf;
typedef __attribute__((ext_vector_type(8)))  float  v8f;

// LDS byte offset of a __shared__ object (addrspace(3) is a 32-bit offset space)
static __device__ __forceinline__ unsigned lds_off_of(const void* p) {
  return (unsigned)(size_t)(__attribute__((address_space(3))) const void*)p;
}

// ---------------------------------------------------------------- FFT helpers
// DIF: natural-order input -> bit-reversed output (forward transform, e^{-i})
__device__ __forceinline__ void fft_dif(float2* sd, int logn, int tid, int nt) {
  const int n = 1 << logn;
  for (int st = logn - 1; st >= 0; --st) {
    const int half = 1 << st;
    for (int i = tid; i < (n >> 1); i += nt) {
      const int grp = i >> st;
      const int pos = i & (half - 1);
      const int idx = (grp << (st + 1)) + pos;
      const float2 u = sd[idx], v = sd[idx + half];
      const float ang = -6.28318530717958647692f * (float)pos / (float)(half << 1);
      float sn, cs; __sincosf(ang, &sn, &cs);
      const float dx = u.x - v.x, dy = u.y - v.y;
      sd[idx]        = make_float2(u.x + v.x, u.y + v.y);
      sd[idx + half] = make_float2(dx * cs - dy * sn, dx * sn + dy * cs);
    }
    __syncthreads();
  }
}

// DIT inverse: bit-reversed input -> natural output (e^{+i}); caller scales 1/N
__device__ __forceinline__ void fft_dit_inv(float2* sd, int logn, int tid, int nt) {
  const int n = 1 << logn;
  for (int st = 0; st < logn; ++st) {
    const int half = 1 << st;
    for (int i = tid; i < (n >> 1); i += nt) {
      const int grp = i >> st;
      const int pos = i & (half - 1);
      const int idx = (grp << (st + 1)) + pos;
      const float ang = 6.28318530717958647692f * (float)pos / (float)(half << 1);
      float sn, cs; __sincosf(ang, &sn, &cs);
      const float2 u = sd[idx], v = sd[idx + half];
      const float tx = v.x * cs - v.y * sn, ty = v.x * sn + v.y * cs;
      sd[idx]        = make_float2(u.x + tx, u.y + ty);
      sd[idx + half] = make_float2(u.x - tx, u.y - ty);
    }
    __syncthreads();
  }
}

// ---------------------------------------------------------------- small math
__device__ __forceinline__ float gelu_exact(float v) {
  return 0.5f * v * (1.0f + erff(v * 0.70710678118654752440f));
}

__device__ __forceinline__ void ln4(float* v, const float* g, const float* b) {
  const float m = 0.25f * (v[0] + v[1] + v[2] + v[3]);
  float var = 0.f;
#pragma unroll
  for (int j = 0; j < 4; ++j) { const float d = v[j] - m; var += d * d; }
  var *= 0.25f;
  const float inv = rsqrtf(var + 1e-5f);
#pragma unroll
  for (int j = 0; j < 4; ++j) v[j] = (v[j] - m) * inv * g[j] + b[j];
}

__device__ __forceinline__ void qnorm(float* q) {
  const float n = sqrtf(q[0]*q[0] + q[1]*q[1] + q[2]*q[2] + q[3]*q[3]) + 1e-8f;
  const float inv = 1.0f / n;
#pragma unroll
  for (int j = 0; j < 4; ++j) q[j] *= inv;
}

__device__ __forceinline__ void qmul(const float* a, const float* b, float* o) {
  o[0] = a[0]*b[0] - a[1]*b[1] - a[2]*b[2] - a[3]*b[3];
  o[1] = a[0]*b[1] + a[1]*b[0] + a[2]*b[3] - a[3]*b[2];
  o[2] = a[0]*b[2] - a[1]*b[3] + a[2]*b[0] + a[3]*b[1];
  o[3] = a[0]*b[3] + a[1]*b[2] - a[2]*b[1] + a[3]*b[0];
}

__device__ __forceinline__ void make_unit_quat(const float* th, float* q) {
  const float c0 = cosf(th[0]*0.5f), c1 = cosf(th[1]*0.5f), c2 = cosf(th[2]*0.5f);
  const float s0 = sinf(th[0]*0.5f), s1 = sinf(th[1]*0.5f), s2 = sinf(th[2]*0.5f);
  q[0] = c0*c1*c2; q[1] = s0*c1*c2; q[2] = c0*s1*c2; q[3] = c0*c1*s2;
  qnorm(q);
}

// ---------------------------------------------------------------- K1: in-proj
__global__ __launch_bounds__(256) void k_inproj(const float* __restrict__ x,
                                                const float* __restrict__ W,
                                                const float* __restrict__ bias,
                                                float* __restrict__ xp) {
  const int row = blockIdx.x;
  const int c4  = threadIdx.x;            // 256 threads * float4 = 1024 cols
  const float4 xv = reinterpret_cast<const float4*>(x)[row];
  const float4* W4 = reinterpret_cast<const float4*>(W);
  const float4 w0 = W4[0*256 + c4], w1 = W4[1*256 + c4];
  const float4 w2 = W4[2*256 + c4], w3 = W4[3*256 + c4];
  const float4 bb = reinterpret_cast<const float4*>(bias)[c4];
  float4 o;
  o.x = bb.x + xv.x*w0.x + xv.y*w1.x + xv.z*w2.x + xv.w*w3.x;
  o.y = bb.y + xv.x*w0.y + xv.y*w1.y + xv.z*w2.y + xv.w*w3.y;
  o.z = bb.z + xv.x*w0.z + xv.y*w1.z + xv.z*w2.z + xv.w*w3.z;
  o.w = bb.w + xv.x*w0.w + xv.y*w1.w + xv.z*w2.w + xv.w*w3.w;
  reinterpret_cast<float4*>(xp)[(size_t)row * 256 + c4] = o;
}

// ---------------------------------------------------------------- K2: gelu_k
__global__ __launch_bounds__(256) void k_gelu_k(float* __restrict__ geluk) {
  __shared__ float2 red[256];
  __shared__ float st[2];
  const int tid = threadIdx.x;
  float s = 0.f, sq = 0.f;
  for (int i = tid; i < SS; i += 256) {
    const float f = (i < SS/2) ? (float)i / SS : (float)(i - SS) / SS;
    const float lk = logf(fabsf(f) + 1e-8f);
    s += lk; sq += lk * lk;
  }
  red[tid] = make_float2(s, sq);
  __syncthreads();
  for (int o = 128; o > 0; o >>= 1) {
    if (tid < o) { red[tid].x += red[tid+o].x; red[tid].y += red[tid+o].y; }
    __syncthreads();
  }
  if (tid == 0) {
    const float m = red[0].x / SS;
    const float var = (red[0].y - (float)SS * m * m) / (float)(SS - 1);  // ddof=1
    st[0] = m; st[1] = 1.0f / (sqrtf(var) + 1e-8f);
  }
  __syncthreads();
  const float m = st[0], inv = st[1];
  for (int i = tid; i < SS; i += 256) {
    const float f = (i < SS/2) ? (float)i / SS : (float)(i - SS) / SS;
    const float lk = logf(fabsf(f) + 1e-8f);
    geluk[i] = gelu_exact((lk - m) * inv);
  }
}

// ------------------------------------------------- K3a: fractal FFT pass 1
__global__ __launch_bounds__(256) void k_fr_pass1(const float* __restrict__ src,
                                                  float2* __restrict__ Y) {
  __shared__ float2 sd[NN1];
  const int tid = threadIdx.x, n1 = blockIdx.x;
  for (int i = tid; i < NN1; i += 256)
    sd[i] = make_float2(src[n1 + (size_t)i * NN1], 0.f);
  __syncthreads();
  fft_dif(sd, LGN1, tid, 256);
  for (int p = tid; p < NN1; p += 256) {
    const int k2 = (int)(__brev((unsigned)p) >> (32 - LGN1));
    const float ang = -6.28318530717958647692f * (float)(n1 * k2) * (1.0f / (float)NFR);
    float sn, cs; __sincosf(ang, &sn, &cs);
    const float2 v = sd[p];
    Y[(size_t)n1 * NN1 + k2] = make_float2(v.x*cs - v.y*sn, v.x*sn + v.y*cs);
  }
}

// ------------------------------------------------- K3b: fractal FFT pass 2
__global__ __launch_bounds__(256) void k_fr_pass2(const float2* __restrict__ Y,
                                                  float* __restrict__ part) {
  __shared__ float2 sd[NN1];
  __shared__ float red[256];
  const int tid = threadIdx.x, k2 = blockIdx.x;
  for (int i = tid; i < NN1; i += 256) sd[i] = Y[(size_t)i * NN1 + k2];
  __syncthreads();
  fft_dif(sd, LGN1, tid, 256);
  float acc[5] = {0.f, 0.f, 0.f, 0.f, 0.f};
  for (int p = tid; p < NN1; p += 256) {
    const int k1 = (int)(__brev((unsigned)p) >> (32 - LGN1));
    const int k  = k2 + (k1 << LGN1);
    const float fr = (float)k * (1.0f / (float)NFR);
    if (fr > 0.01f && fr < 0.5f) {
      const float2 v = sd[p];
      const float lp = logf(v.x*v.x + v.y*v.y + 1e-10f);
      const float lk = logf(fr + 1e-10f);
      acc[0] += 1.f; acc[1] += lk; acc[2] += lk*lk; acc[3] += lp; acc[4] += lk*lp;
    }
  }
  __syncthreads();
  for (int q = 0; q < 5; ++q) {
    red[tid] = acc[q];
    __syncthreads();
    for (int o = 128; o > 0; o >>= 1) {
      if (tid < o) red[tid] += red[tid + o];
      __syncthreads();
    }
    if (tid == 0) part[(size_t)k2 * 5 + q] = red[0];
    __syncthreads();
  }
}

// ------------------------------------------------- K3c: alpha per batch
__global__ __launch_bounds__(256) void k_alpha(const float* __restrict__ part,
                                               float* __restrict__ alpha) {
  __shared__ float red[256];
  const int tid = threadIdx.x;
  for (int b = 0; b < BB; ++b) {
    float acc[5] = {0.f, 0.f, 0.f, 0.f, 0.f};
    for (int r = tid; r < NN1; r += 256) {
      const float* p = part + ((size_t)b * NN1 + r) * 5;
#pragma unroll
      for (int q = 0; q < 5; ++q) acc[q] += p[q];
    }
    float tot[5];
    for (int q = 0; q < 5; ++q) {
      red[tid] = acc[q];
      __syncthreads();
      for (int o = 128; o > 0; o >>= 1) {
        if (tid < o) red[tid] += red[tid + o];
        __syncthreads();
      }
      tot[q] = red[0];
      __syncthreads();
    }
    if (tid == 0) {
      const float n = tot[0], K1 = tot[1], K2 = tot[2], A = tot[3], C = tot[4];
      const float mk  = K1 / n;
      const float var = K2 - K1 * K1 / n;      // sum w*dk*dk
      const float cov = C - mk * A;            // sum w*dk*logP
      const float beta = -cov / (var + 1e-10f);
      const float D = fminf(fmaxf((3.0f - beta) * 0.5f, 0.5f), 1.5f);
      alpha[b] = fminf(fmaxf(1.0f + 0.8f * (D - 1.0f), 0.1f), 3.0f);
    }
    __syncthreads();
  }
}

// ------------------------------------------------- K4: spectral filter
// Emits bf16 (A-operand of the WMMA GEMM) directly.
__global__ __launch_bounds__(256) void k_filter(const float* __restrict__ xp,
                                                __bf16* __restrict__ xf,
                                                const float* __restrict__ geluk,
                                                const float* __restrict__ alpha,
                                                const float* __restrict__ abase,
                                                const float* __restrict__ pshift) {
  __shared__ float2 sd[SS];
  const int tid = threadIdx.x;
  const int idx = blockIdx.x;
  const int b   = idx >> 10;          // H*HD = 1024
  const int col = idx & 1023;         // h*64 + d  (column in DM)
  const int h   = col >> 6;
  const float* src = xp + ((size_t)b * SS) * DMM + col;
  for (int i = tid; i < SS; i += 256) sd[i] = make_float2(src[(size_t)i * DMM], 0.f);
  __syncthreads();
  fft_dif(sd, LGS, tid, 256);
  const float sc  = abase[h] * alpha[b];
  const float psh = pshift[h];
  for (int p = tid; p < SS; p += 256) {
    const int k = (int)(__brev((unsigned)p) >> (32 - LGS));   // natural bin
    const float ph = sc * geluk[k] + psh;
    float sn, cs; __sincosf(ph, &sn, &cs);
    const float2 v = sd[p];
    sd[p] = make_float2(v.x*cs - v.y*sn, v.x*sn + v.y*cs);
  }
  __syncthreads();
  fft_dit_inv(sd, LGS, tid, 256);
  __bf16* dst = xf + ((size_t)b * SS) * DMM + col;
  const float inv = 1.0f / (float)SS;
  for (int i = tid; i < SS; i += 256) dst[(size_t)i * DMM] = (__bf16)(sd[i].x * inv);
}

// ------------------------------------------------- K4b: W -> bf16, transposed
// Wt[n][k] = bf16(W[k][n]); tiled through LDS for coalescing.
__global__ __launch_bounds__(256) void k_prep_w(const float* __restrict__ W,
                                                __bf16* __restrict__ Wt) {
  __shared__ float tile[32][33];
  const int k0 = blockIdx.y * 32, n0 = blockIdx.x * 32;
  const int tx = threadIdx.x & 31, ty = threadIdx.x >> 5;   // 8 rows per pass
  for (int r = ty; r < 32; r += 8)
    tile[r][tx] = W[(size_t)(k0 + r) * DMM + n0 + tx];
  __syncthreads();
  for (int r = ty; r < 32; r += 8)
    Wt[(size_t)(n0 + r) * DMM + k0 + tx] = (__bf16)tile[tx][r];
}

// ------------------------------------------------- K5: big GEMM (WMMA bf16)
// C[32768,1024] = A[32768,1024](bf16) * Wt^T + bias.
// Block tile 128x128, 8 waves (4x2), wave tile 32x64 = 8 WMMAs per K-step.
// LDS tiles filled with global_load_async_to_lds_b128 (ASYNCcnt path).
__device__ __forceinline__ v16bf load_frag16(const __bf16* base, int ld,
                                             int row0, int lane) {
  // 16-bit A-layout (ISA 7.12.2): lanes 0-15 -> M/N=row0+lane, K-group 0;
  // lanes 16-31 -> row0+lane-16, K-group 8.  VGPR j holds K pairs.
  const int m  = row0 + (lane & 15);
  const int kb = (lane >> 4) << 3;
  const __bf16* p = base + m * ld;
  v16bf f;
#pragma unroll
  for (int j = 0; j < 8; ++j) {
    f[j]     = p[kb + j];           // K = kb + j
    f[j + 8] = p[16 + kb + j];      // K = 16 + kb + j
  }
  return f;
}

__global__ __launch_bounds__(256) void k_gemm(const __bf16* __restrict__ Abf,
                                              const __bf16* __restrict__ Wt,
                                              const float* __restrict__ bias,
                                              float* __restrict__ Cg) {
  __shared__ __align__(16) __bf16 Asl[128][40];   // [M][K], 80B row stride
  __shared__ __align__(16) __bf16 Bsl[128][40];   // [N][K]
  const int tid  = threadIdx.x;
  const int lane = tid & 31;
  const int wave = tid >> 5;
  const int wm   = wave >> 1;       // 0..3 -> M offset 32*wm
  const int wn   = wave & 1;        // 0..1 -> N offset 64*wn
  const int m0   = blockIdx.y * 128;
  const int n0   = blockIdx.x * 128;

  v8f acc[2][4];
#pragma unroll
  for (int a = 0; a < 2; ++a)
#pragma unroll
    for (int c = 0; c < 4; ++c) acc[a][c] = {};

  for (int kk = 0; kk < DMM; kk += 32) {
    // async-stage A tile (128x32 bf16 = 512 x 16B chunks, 2 per thread)
#pragma unroll
    for (int rep = 0; rep < 2; ++rep) {
      const int ch = tid + rep * 256;
      const int r = ch >> 2, c = (ch & 3) << 3;
      const unsigned       lds = lds_off_of(&Asl[r][c]);
      const unsigned long long ga =
          (unsigned long long)(Abf + (size_t)(m0 + r) * DMM + kk + c);
      asm volatile("global_load_async_to_lds_b128 %0, %1, off"
                   :: "v"(lds), "v"(ga) : "memory");
    }
    // async-stage B tile (128x32 bf16, from pre-transposed Wt[N][K])
#pragma unroll
    for (int rep = 0; rep < 2; ++rep) {
      const int ch = tid + rep * 256;
      const int r = ch >> 2, c = (ch & 3) << 3;
      const unsigned       lds = lds_off_of(&Bsl[r][c]);
      const unsigned long long ga =
          (unsigned long long)(Wt + (size_t)(n0 + r) * DMM + kk + c);
      asm volatile("global_load_async_to_lds_b128 %0, %1, off"
                   :: "v"(lds), "v"(ga) : "memory");
    }
    asm volatile("s_wait_asynccnt 0x0" ::: "memory");
    __syncthreads();

    v16bf fa[2], fb[4];
#pragma unroll
    for (int a = 0; a < 2; ++a)
      fa[a] = load_frag16(&Asl[0][0], 40, wm * 32 + a * 16, lane);
#pragma unroll
    for (int c = 0; c < 4; ++c)
      fb[c] = load_frag16(&Bsl[0][0], 40, wn * 64 + c * 16, lane);
#pragma unroll
    for (int a = 0; a < 2; ++a)
#pragma unroll
      for (int c = 0; c < 4; ++c)
        acc[a][c] = __builtin_amdgcn_wmma_f32_16x16x32_bf16(
            false, fa[a], false, fb[c], (short)0, acc[a][c], false, false);
    __syncthreads();
  }

  // epilogue per ISA C/D layout: lanes 0-15 -> N=lane, VGPR r -> M=r;
  //                              lanes 16-31 -> N=lane-16, M=r+8
  const int mrow0 = m0 + wm * 32 + ((lane < 16) ? 0 : 8);
  const int ncol0 = n0 + wn * 64 + (lane & 15);
#pragma unroll
  for (int a = 0; a < 2; ++a) {
#pragma unroll
    for (int c = 0; c < 4; ++c) {
      const int ncol = ncol0 + c * 16;
      const float bn = bias[ncol];
#pragma unroll
      for (int r = 0; r < 8; ++r)
        Cg[(size_t)(mrow0 + a * 16 + r) * DMM + ncol] = acc[a][c][r] + bn;
    }
  }
}

// ------------------------------------------------- K6: residual + LayerNorm
__global__ __launch_bounds__(256) void k_ln_res(const float* __restrict__ xp,
                                                float* __restrict__ io,
                                                const float* __restrict__ g,
                                                const float* __restrict__ bb) {
  __shared__ float2 red[256];
  __shared__ float st[2];
  const size_t base = (size_t)blockIdx.x * DMM;
  float v[4];
  float s = 0.f, sq = 0.f;
#pragma unroll
  for (int j = 0; j < 4; ++j) {
    const int c = threadIdx.x + j * 256;
    const float t = xp[base + c] + io[base + c];
    v[j] = t; s += t; sq += t * t;
  }
  red[threadIdx.x] = make_float2(s, sq);
  __syncthreads();
  for (int o = 128; o > 0; o >>= 1) {
    if (threadIdx.x < o) {
      red[threadIdx.x].x += red[threadIdx.x + o].x;
      red[threadIdx.x].y += red[threadIdx.x + o].y;
    }
    __syncthreads();
  }
  if (threadIdx.x == 0) {
    const float m = red[0].x / DMM;
    st[0] = m;
    st[1] = rsqrtf(red[0].y / DMM - m * m + 1e-5f);
  }
  __syncthreads();
  const float m = st[0], inv = st[1];
#pragma unroll
  for (int j = 0; j < 4; ++j) {
    const int c = threadIdx.x + j * 256;
    io[base + c] = (v[j] - m) * inv * g[c] + bb[c];
  }
}

// ------------------------------------------------- K7: out-proj + quat tail
__global__ __launch_bounds__(256) void k_tail(const float* __restrict__ attn,
                                              const float* __restrict__ x,
                                              const float* __restrict__ W_out,
                                              const float* __restrict__ b_out,
                                              const float* __restrict__ thL,
                                              const float* __restrict__ thR,
                                              const float* __restrict__ Wf1,
                                              const float* __restrict__ bf1,
                                              const float* __restrict__ Wf2,
                                              const float* __restrict__ bf2,
                                              const float* __restrict__ n1g,
                                              const float* __restrict__ n1b,
                                              const float* __restrict__ n2g,
                                              const float* __restrict__ n2b,
                                              const float* __restrict__ n3g,
                                              const float* __restrict__ n3b,
                                              float* __restrict__ out) {
  const int lane = threadIdx.x & 31;
  const int row  = blockIdx.x * 8 + (threadIdx.x >> 5);
  const float* ar = attn + (size_t)row * DMM;
  float a0 = 0.f, a1 = 0.f, a2 = 0.f, a3 = 0.f;
  for (int k = lane; k < DMM; k += 32) {
    const float a = ar[k];
    const float4 w = reinterpret_cast<const float4*>(W_out)[k];
    a0 += a * w.x; a1 += a * w.y; a2 += a * w.z; a3 += a * w.w;
  }
#pragma unroll
  for (int o = 16; o > 0; o >>= 1) {
    a0 += __shfl_xor(a0, o, 32);
    a1 += __shfl_xor(a1, o, 32);
    a2 += __shfl_xor(a2, o, 32);
    a3 += __shfl_xor(a3, o, 32);
  }
  if (lane != 0) return;

  float h[4];
  const float4 xv = reinterpret_cast<const float4*>(x)[row];
  h[0] = xv.x + a0 + b_out[0];
  h[1] = xv.y + a1 + b_out[1];
  h[2] = xv.z + a2 + b_out[2];
  h[3] = xv.w + a3 + b_out[3];
  ln4(h, n1g, n1b); qnorm(h);

  float ev[4] = {h[0], h[1], h[2], h[3]};
#pragma unroll
  for (int i = 0; i < NROT; ++i) {
    float qL[4], qR[4], t1[4];
    make_unit_quat(&thL[i * 3], qL);
    make_unit_quat(&thR[i * 3], qR);
    qR[1] = -qR[1]; qR[2] = -qR[2]; qR[3] = -qR[3];   // conjugate
    qmul(qL, ev, t1);
    qmul(t1, qR, ev);
    qnorm(ev);
  }

  float h2[4];
#pragma unroll
  for (int j = 0; j < 4; ++j) h2[j] = h[j] + ev[j];
  ln4(h2, n2g, n2b); qnorm(h2);

  float z[16];
#pragma unroll
  for (int j = 0; j < 16; ++j) {
    float t = bf1[j];
#pragma unroll
    for (int i = 0; i < 4; ++i) t += h2[i] * Wf1[i * 16 + j];
    z[j] = gelu_exact(t);
  }
  float y[4];
#pragma unroll
  for (int i = 0; i < 4; ++i) {
    float t = bf2[i];
#pragma unroll
    for (int j = 0; j < 16; ++j) t += z[j] * Wf2[j * 4 + i];
    y[i] = h2[i] + t;
  }
  ln4(y, n3g, n3b); qnorm(y);
  reinterpret_cast<float4*>(out)[row] = make_float4(y[0], y[1], y[2], y[3]);
}

// ---------------------------------------------------------------- launcher
extern "C" void kernel_launch(void* const* d_in, const int* in_sizes, int n_in,
                              void* d_out, int out_size, void* d_ws, size_t ws_size,
                              hipStream_t stream) {
  (void)in_sizes; (void)n_in; (void)out_size; (void)ws_size;
  const float* x          = (const float*)d_in[0];
  const float* W_in       = (const float*)d_in[1];
  const float* b_in       = (const float*)d_in[2];
  const float* alpha_base = (const float*)d_in[3];
  const float* phase_sh   = (const float*)d_in[4];
  const float* W_attn_out = (const float*)d_in[5];
  const float* b_attn_out = (const float*)d_in[6];
  const float* ln_attn_g  = (const float*)d_in[7];
  const float* ln_attn_b  = (const float*)d_in[8];
  const float* W_out      = (const float*)d_in[9];
  const float* b_out      = (const float*)d_in[10];
  const float* theta_L    = (const float*)d_in[11];
  const float* theta_R    = (const float*)d_in[12];
  const float* W_ff1      = (const float*)d_in[13];
  const float* b_ff1      = (const float*)d_in[14];
  const float* W_ff2      = (const float*)d_in[15];
  const float* b_ff2      = (const float*)d_in[16];
  const float* n1_g       = (const float*)d_in[17];
  const float* n1_b       = (const float*)d_in[18];
  const float* n2_g       = (const float*)d_in[19];
  const float* n2_b       = (const float*)d_in[20];
  const float* n3_g       = (const float*)d_in[21];
  const float* n3_b       = (const float*)d_in[22];
  float* out = (float*)d_out;

  // workspace carve-up (float units; all offsets 16B-aligned)
  float* ws = (float*)d_ws;
  size_t off = 0;
  float*  xproj  = ws + off; off += (size_t)ROWS * DMM;          // 134 MB
  float*  gout   = ws + off; off += (size_t)ROWS * DMM;          // 134 MB (becomes attn)
  __bf16* xfiltb = (__bf16*)(ws + off); off += (size_t)ROWS * DMM / 2;   // 67 MB
  __bf16* Wtb    = (__bf16*)(ws + off); off += (size_t)DMM * DMM / 2;    // 2 MB
  float2* Y      = (float2*)(ws + off); off += (size_t)NFR * 2;  // 33.5 MB, reused/batch
  float*  partials = ws + off; off += (size_t)BB * NN1 * 5;
  float*  geluk    = ws + off; off += SS;
  float*  alphab   = ws + off; off += 16;

  // 1. input projection
  k_inproj<<<ROWS, 256, 0, stream>>>(x, W_in, b_in, xproj);
  // 2. gelu(log-k) table + bf16 transposed weight
  k_gelu_k<<<1, 256, 0, stream>>>(geluk);
  k_prep_w<<<dim3(32, 32), 256, 0, stream>>>(W_attn_out, Wtb);
  // 3. fractal_alpha: four-step 2^22-point FFT per batch (Y reused, serialized)
  for (int b = 0; b < BB; ++b) {
    k_fr_pass1<<<NN1, 256, 0, stream>>>(xproj + (size_t)b * NFR, Y);
    k_fr_pass2<<<NN1, 256, 0, stream>>>(Y, partials + (size_t)b * NN1 * 5);
  }
  k_alpha<<<1, 256, 0, stream>>>(partials, alphab);
  // 4. spectral filter: fwd FFT + phase + inv FFT per (b,h,d) -> bf16 A-matrix
  k_filter<<<BB * HH * HDD, 256, 0, stream>>>(xproj, xfiltb, geluk, alphab,
                                              alpha_base, phase_sh);
  // 5. out-projection GEMM (bf16 WMMA, async-to-LDS staging) + bias
  k_gemm<<<dim3(DMM / 128, ROWS / 128), 256, 0, stream>>>(xfiltb, Wtb,
                                                          b_attn_out, gout);
  // 6. residual + LayerNorm  -> attn (in place in gout)
  k_ln_res<<<ROWS, 256, 0, stream>>>(xproj, gout, ln_attn_g, ln_attn_b);
  // 7. attn @ W_out + quaternion / SO(4) / FF tail, fused per row
  k_tail<<<ROWS / 8, 256, 0, stream>>>(gout, x, W_out, b_out, theta_L, theta_R,
                                       W_ff1, b_ff1, W_ff2, b_ff2,
                                       n1_g, n1_b, n2_g, n2_b, n3_g, n3_b, out);
}
